// Feature_extractor_38208029065288
// MI455X (gfx1250) — matrix operand order repaired and checked by reading, not compiled, not run on hardware
//
#include <hip/hip_runtime.h>

// ---------------------------------------------------------------------------
// MI455X (gfx1250): GAT x4 (shared weights) + 2-layer LSTM.
// GEMMs on v_wmma_f32_16x16x32_bf16, double-buffered LDS, async global->LDS
// staging for the A operand (CDNA5 Tensor/async path), software-pipelined B.
// ---------------------------------------------------------------------------

typedef __attribute__((ext_vector_type(16))) __bf16 v16bf;
typedef __attribute__((ext_vector_type(8)))  float  v8f;

// Flat address of an LDS object: low 32 bits == LDS byte offset (ISA 10.2).
__device__ __forceinline__ unsigned lds_off(const void* p) {
  return (unsigned)(unsigned long long)p;
}

// ---------------- elementwise / prep kernels ----------------

__global__ void k_cvt_bf16(const float* __restrict__ s, __bf16* __restrict__ d, long n) {
  long i = (long)blockIdx.x * blockDim.x + threadIdx.x;
  if (i < n) d[i] = (__bf16)s[i];
}

// src f32 [R,C] row-major -> dst bf16 [C,R] row-major (weight [out,in] -> [K,N])
__global__ void k_transpose_bf16(const float* __restrict__ s, __bf16* __restrict__ d,
                                 int R, int C) {
  long i = (long)blockIdx.x * blockDim.x + threadIdx.x;
  if (i < (long)R * C) {
    int r = (int)(i / C), c = (int)(i % C);
    d[(long)c * R + r] = (__bf16)s[i];
  }
}

__global__ void k_zero_bf16(__bf16* __restrict__ d, long n) {
  long i = (long)blockIdx.x * blockDim.x + threadIdx.x;
  if (i < n) d[i] = (__bf16)0.0f;
}

// ---------------- WMMA GEMM ----------------
// C[M,N] (+)= A[M,K] * B[K,N] (+ bias[n]); A,B bf16 row-major.
// Block tile 128x128, K-step 32, 8 waves, wave tile 32x64 (2x4 WMMA subtiles).
// A staged global->LDS via global_load_async_to_lds_b128 (layout matches frag),
// B register-staged + transposed scatter into Bs[n][k]. Double-buffered LDS.
// Requires: M%128==0, N%128==0, K%32==0, lda/ldb and col offsets %8==0.

__global__ __launch_bounds__(256)
void k_gemm_bf16(const __bf16* __restrict__ A, int lda, long sA,
                 const __bf16* __restrict__ B, int ldb, long sB,
                 float* __restrict__ Cf, __bf16* __restrict__ Cb, int ldc, long sC,
                 const float* __restrict__ bias, int accumulate,
                 int Md, int Nd, int Kd) {
  (void)Md; (void)Nd;
  __shared__ __bf16 As[2][128 * 32];        // [m][k], 8 KB per buffer
  __shared__ __bf16 Bs[2][128 * 32];        // [n][k] transposed, 8 KB per buffer

  A += (long)blockIdx.z * sA;
  B += (long)blockIdx.z * sB;
  const long cOff = (long)blockIdx.z * sC;

  const int blockRow = blockIdx.y * 128;
  const int blockCol = blockIdx.x * 128;
  const int tid  = threadIdx.x;
  const int lane = tid & 31;
  const int wave = tid >> 5;
  const int wm = wave & 3;                  // 4 row groups of 32
  const int wn = wave >> 2;                 // 2 col groups of 64
  const int half = lane >> 4;
  const int lr   = lane & 15;

  union Frag { uint4 u[2]; v16bf v; };
  v8f acc[2][4] = {};

  const unsigned asBase0 = lds_off(&As[0][0]);
  const unsigned asBase1 = lds_off(&As[1][0]);

  // --- A staging: per thread 2 async b128 copies, LDS layout == memory layout
  auto stageA = [&](int kk, int buf) {
    const int k0 = kk << 5;
    const unsigned base = buf ? asBase1 : asBase0;
#pragma unroll
    for (int j = 0; j < 2; ++j) {
      int idx = tid + j * 256;              // 0..511 (16B chunks)
      int m   = idx >> 2;                   // 0..127
      int ko  = (idx & 3) << 3;             // 0,8,16,24
      unsigned long long g =
          (unsigned long long)(const void*)(A + (long)(blockRow + m) * lda + k0 + ko);
      unsigned l = base + (unsigned)idx * 16u;
      asm volatile("global_load_async_to_lds_b128 %0, %1, off"
                   :: "v"(l), "v"(g) : "memory");
    }
  };

  // --- B staging: load 2x16B into regs early, scatter-transpose after compute
  uint4 bReg[2];
  auto loadB = [&](int kk) {
    const int k0 = kk << 5;
#pragma unroll
    for (int j = 0; j < 2; ++j) {
      int idx = tid + j * 256;              // 0..511
      int kr  = idx >> 4;                   // 0..31
      int n0  = (idx & 15) << 3;            // 0..120
      bReg[j] = *reinterpret_cast<const uint4*>(B + (long)(k0 + kr) * ldb + blockCol + n0);
    }
  };
  auto scatterB = [&](int buf) {
#pragma unroll
    for (int j = 0; j < 2; ++j) {
      int idx = tid + j * 256;
      int kr  = idx >> 4;
      int n0  = (idx & 15) << 3;
      union { uint4 u; __bf16 e[8]; } bu;
      bu.u = bReg[j];
#pragma unroll
      for (int e = 0; e < 8; ++e) Bs[buf][(n0 + e) * 32 + kr] = bu.e[e];
    }
  };

  // --- compute: 2 A frags x 4 B frags -> 8 WMMAs
  auto compute = [&](int buf) {
    Frag a[2], b[4];
#pragma unroll
    for (int i = 0; i < 2; ++i) {
      const __bf16* base = &As[buf][(wm * 32 + i * 16 + lr) * 32];
      a[i].u[0] = *reinterpret_cast<const uint4*>(base + half * 8);
      a[i].u[1] = *reinterpret_cast<const uint4*>(base + 16 + half * 8);
    }
#pragma unroll
    for (int j = 0; j < 4; ++j) {
      const __bf16* base = &Bs[buf][(wn * 64 + j * 16 + lr) * 32];
      b[j].u[0] = *reinterpret_cast<const uint4*>(base + half * 8);
      b[j].u[1] = *reinterpret_cast<const uint4*>(base + 16 + half * 8);
    }
#pragma unroll
    for (int i = 0; i < 2; ++i)
#pragma unroll
      for (int j = 0; j < 4; ++j)
        acc[i][j] = __builtin_amdgcn_wmma_f32_16x16x32_bf16(
            false, a[i].v, false, b[j].v, (short)0, acc[i][j], false, false);
  };

  // --- pipelined main loop
  stageA(0, 0);
  loadB(0);
  asm volatile("s_wait_asynccnt 0" ::: "memory");
  scatterB(0);
  __syncthreads();

  const int ksteps = Kd >> 5;
  for (int kk = 0; kk < ksteps; ++kk) {
    const int cur = kk & 1, nxt = cur ^ 1;
    const bool more = (kk + 1) < ksteps;
    if (more) { stageA(kk + 1, nxt); loadB(kk + 1); }
    compute(cur);
    if (more) {
      scatterB(nxt);
      asm volatile("s_wait_asynccnt 0" ::: "memory");
    }
    __syncthreads();
  }

  // --- epilogue: C/D f32 layout: VGPR r -> M=r(+8 for lanes 16..31), N=lr
#pragma unroll
  for (int i = 0; i < 2; ++i) {
#pragma unroll
    for (int j = 0; j < 4; ++j) {
      int n = blockCol + wn * 64 + j * 16 + lr;
      float bv = bias ? bias[n] : 0.0f;
      int m0 = blockRow + wm * 32 + i * 16 + half * 8;
#pragma unroll
      for (int r = 0; r < 8; ++r) {
        long idx = cOff + (long)(m0 + r) * ldc + n;
        float v = acc[i][j][r] + bv;
        if (accumulate) v += Cf[idx];
        if (Cf) Cf[idx] = v;
        if (Cb) Cb[idx] = (__bf16)v;
      }
    }
  }
}

// ---------------- GAT attention pieces ----------------

// src[h,i] = sum_c h[i,h,c]*a[h,c]; dst[h,i] = sum_c h[i,h,c]*a[h,128+c]
__global__ void k_srcdst(const float* __restrict__ hmat, const float* __restrict__ aattn,
                         float* __restrict__ srcv, float* __restrict__ dstv,
                         int Nn, int Dd, int Hc) {
  int wid  = blockIdx.x * 8 + (threadIdx.x >> 5);
  int lane = threadIdx.x & 31;
  int h = wid & 7;
  int i = wid >> 3;
  const float* hp = hmat + (long)i * Dd + h * Hc + lane * 4;
  const float* a1 = aattn + h * 2 * Hc + lane * 4;
  const float* a2 = a1 + Hc;
  float s = 0.0f, d = 0.0f;
#pragma unroll
  for (int q = 0; q < 4; ++q) { float hv = hp[q]; s += hv * a1[q]; d += hv * a2[q]; }
#pragma unroll
  for (int o = 16; o; o >>= 1) { s += __shfl_xor(s, o, 32); d += __shfl_xor(d, o, 32); }
  if (lane == 0) { srcv[(long)h * Nn + i] = s; dstv[(long)h * Nn + i] = d; }
}

// Row softmax over j of masked leakyrelu(src+dst); writes f32 probs (= final attn
// output layout [H,N,N]) and bf16 probs for the WMMA aggregation GEMM.
__global__ void k_softmax(const float* __restrict__ srcv, const float* __restrict__ dstv,
                          const float* __restrict__ adj, float* __restrict__ pF,
                          __bf16* __restrict__ pB, int Nn) {
  int lane = threadIdx.x & 31;
  int i = blockIdx.x * 8 + (threadIdx.x >> 5);
  int h = blockIdx.y;
  float s = srcv[(long)h * Nn + i];
  const float* dro = dstv + (long)h * Nn;
  const float* aro = adj + (long)i * Nn;

  float mx = -3.0e38f;
  for (int j = lane; j < Nn; j += 32) {
    float l = s + dro[j]; l = l > 0.0f ? l : 0.2f * l;
    float ml = aro[j] > 0.0f ? l : -9.0e15f;
    mx = fmaxf(mx, ml);
  }
#pragma unroll
  for (int o = 16; o; o >>= 1) mx = fmaxf(mx, __shfl_xor(mx, o, 32));

  float sum = 0.0f;
  for (int j = lane; j < Nn; j += 32) {
    float l = s + dro[j]; l = l > 0.0f ? l : 0.2f * l;
    float ml = aro[j] > 0.0f ? l : -9.0e15f;
    sum += __expf(ml - mx);
  }
#pragma unroll
  for (int o = 16; o; o >>= 1) sum += __shfl_xor(sum, o, 32);
  float inv = 1.0f / sum;

  long rowo = ((long)h * Nn + i) * Nn;
  for (int j = lane; j < Nn; j += 32) {
    float l = s + dro[j]; l = l > 0.0f ? l : 0.2f * l;
    float ml = aro[j] > 0.0f ? l : -9.0e15f;
    float p = __expf(ml - mx) * inv;
    pF[rowo + j] = p;
    pB[rowo + j] = (__bf16)p;
  }
}

// ---------------- LSTM cell (PyTorch gate order i,f,g,o) ----------------

__global__ void k_cell(const float* __restrict__ gates, float* __restrict__ c,
                       __bf16* __restrict__ hbf, __bf16* __restrict__ seqstore,
                       float* __restrict__ outp, float scale, int first, int Dh) {
  long i = (long)blockIdx.x * blockDim.x + threadIdx.x;
  long n = i >> 10;                 // Dh == 1024
  int  d = (int)(i & (Dh - 1));
  long base = n * (4L * Dh) + d;
  float gi = gates[base];
  float gf = gates[base + Dh];
  float gg = gates[base + 2 * Dh];
  float go = gates[base + 3 * Dh];
  float cp = first ? 0.0f : c[i];
  auto sig = [](float x) { return 1.0f / (1.0f + __expf(-x)); };
  float cn = sig(gf) * cp + sig(gi) * tanhf(gg);
  float hv = sig(go) * tanhf(cn);
  c[i] = cn;
  hbf[i] = (__bf16)hv;
  if (seqstore) seqstore[i] = (__bf16)hv;
  if (outp) { float v = hv * scale; outp[i] = first ? v : outp[i] + v; }
}

// ---------------- host orchestration ----------------

extern "C" void kernel_launch(void* const* d_in, const int* in_sizes, int n_in,
                              void* d_out, int out_size, void* d_ws, size_t ws_size,
                              hipStream_t stream) {
  (void)in_sizes; (void)n_in; (void)out_size; (void)ws_size;
  const float* patch = (const float*)d_in[0];
  const float* adj   = (const float*)d_in[1];
  const float* Wproj = (const float*)d_in[2];
  const float* bproj = (const float*)d_in[3];
  const float* Wgat  = (const float*)d_in[4];
  const float* bgat  = (const float*)d_in[5];
  const float* aattn = (const float*)d_in[6];
  const float* wih0  = (const float*)d_in[7];
  const float* whh0  = (const float*)d_in[8];
  const float* bih0  = (const float*)d_in[9];
  const float* bhh0  = (const float*)d_in[10];
  const float* wih1  = (const float*)d_in[11];
  const float* whh1  = (const float*)d_in[12];
  const float* bih1  = (const float*)d_in[13];
  const float* bhh1  = (const float*)d_in[14];

  const int N = 2048, FT = 2048, D = 1024, H = 8, HC = 128, G = 4096;
  const long ND = (long)N * D;

  float* outp  = (float*)d_out;          // [N, D]
  float* attnp = outp + ND;              // [H, N, N] (final-layer probs)

  size_t off = 0;
  auto alloc = [&](size_t bytes) {
    void* p = (char*)d_ws + off;
    off += (bytes + 255) & ~(size_t)255;
    return p;
  };
  __bf16* pf_bf    = (__bf16*)alloc((size_t)N * FT * 2);
  __bf16* WpT      = (__bf16*)alloc((size_t)FT * D * 2);   // [FT, D]
  __bf16* WgT      = (__bf16*)alloc((size_t)D * D * 2);    // [D, D]
  __bf16* wih0T    = (__bf16*)alloc((size_t)D * G * 2);    // [D, 4D]
  __bf16* whh0T    = (__bf16*)alloc((size_t)D * G * 2);
  __bf16* wih1T    = (__bf16*)alloc((size_t)D * G * 2);
  __bf16* whh1T    = (__bf16*)alloc((size_t)D * G * 2);
  __bf16* x_bf     = (__bf16*)alloc((size_t)ND * 2);
  __bf16* h_bf     = (__bf16*)alloc((size_t)ND * 2);
  float*  h_f32    = (float*) alloc((size_t)ND * 4);
  float*  srcv     = (float*) alloc((size_t)H * N * 4);
  float*  dstv     = (float*) alloc((size_t)H * N * 4);
  __bf16* probs_bf = (__bf16*)alloc((size_t)H * N * N * 2);
  __bf16* seq_bf   = (__bf16*)alloc((size_t)4 * ND * 2);
  __bf16* seq1_bf  = (__bf16*)alloc((size_t)4 * ND * 2);
  float*  gates    = (float*) alloc((size_t)N * G * 4);
  float*  c_f32    = (float*) alloc((size_t)ND * 4);

  auto gemm = [&](const __bf16* A, int lda, long sA, const __bf16* B, int ldb, long sB,
                  float* Cf, __bf16* Cb, int ldc, long sC, const float* bias, int acc,
                  int Md, int Nd, int Kd, int Z) {
    dim3 g(Nd / 128, Md / 128, Z);
    k_gemm_bf16<<<g, 256, 0, stream>>>(A, lda, sA, B, ldb, sB, Cf, Cb, ldc, sC,
                                       bias, acc, Md, Nd, Kd);
  };

  // ---- prep: bf16 conversions + weight transposes ([out,in] -> [K=in, N=out])
  k_cvt_bf16<<<(unsigned)(((long)N * FT + 255) / 256), 256, 0, stream>>>(patch, pf_bf, (long)N * FT);
  k_transpose_bf16<<<(unsigned)(((long)D * FT + 255) / 256), 256, 0, stream>>>(Wproj, WpT, D, FT);
  k_transpose_bf16<<<(unsigned)(((long)D * D + 255) / 256), 256, 0, stream>>>(Wgat, WgT, D, D);
  k_transpose_bf16<<<(unsigned)(((long)G * D + 255) / 256), 256, 0, stream>>>(wih0, wih0T, G, D);
  k_transpose_bf16<<<(unsigned)(((long)G * D + 255) / 256), 256, 0, stream>>>(whh0, whh0T, G, D);
  k_transpose_bf16<<<(unsigned)(((long)G * D + 255) / 256), 256, 0, stream>>>(wih1, wih1T, G, D);
  k_transpose_bf16<<<(unsigned)(((long)G * D + 255) / 256), 256, 0, stream>>>(whh1, whh1T, G, D);

  // ---- projection: x = patch @ Wproj^T + b  -> bf16 only
  gemm(pf_bf, FT, 0, WpT, D, 0, nullptr, x_bf, D, 0, bproj, 0, N, D, FT, 1);

  // ---- 4 GAT layers (shared weights); layer output goes straight into seq slot
  const __bf16* xcur = x_bf;
  for (int l = 0; l < 4; ++l) {
    // h = x @ Wgat^T + b  (f32 for attention math, bf16 for aggregation B operand)
    gemm(xcur, D, 0, WgT, D, 0, h_f32, h_bf, D, 0, bgat, 0, N, D, D, 1);
    k_srcdst<<<N * H / 8, 256, 0, stream>>>(h_f32, aattn, srcv, dstv, N, D, HC);
    dim3 gs(N / 8, H);
    k_softmax<<<gs, 256, 0, stream>>>(srcv, dstv, adj, attnp, probs_bf, N);
    // out[i,h,c] = sum_j probs[h,i,j] * h[j,h,c]  (batched over heads, grid.z = 8)
    __bf16* xout = seq_bf + (long)l * ND;
    gemm(probs_bf, N, (long)N * N, h_bf, D, (long)HC,
         nullptr, xout, D, (long)HC, nullptr, 0, N, HC, N, H);
    xcur = xout;
  }

  // ---- 2-layer LSTM over seq (T=4); layer 2 accumulates mean into d_out
  auto lstm = [&](const __bf16* seqin, const __bf16* TW_ih, const __bf16* TW_hh,
                  const float* bi, const float* bh, __bf16* seqout, float* finalOut) {
    k_zero_bf16<<<(unsigned)((ND + 255) / 256), 256, 0, stream>>>(h_bf, ND);
    for (int t = 0; t < 4; ++t) {
      gemm(seqin + (long)t * ND, D, 0, TW_ih, G, 0, gates, nullptr, G, 0, bi, 0, N, G, D, 1);
      gemm(h_bf, D, 0, TW_hh, G, 0, gates, nullptr, G, 0, bh, 1, N, G, D, 1);
      k_cell<<<(unsigned)(ND / 256), 256, 0, stream>>>(
          gates, c_f32, h_bf,
          seqout ? seqout + (long)t * ND : nullptr,
          finalOut, 0.25f, t == 0 ? 1 : 0, D);
    }
  };
  lstm(seq_bf,  wih0T, whh0T, bih0, bhh0, seq1_bf, nullptr);
  lstm(seq1_bf, wih1T, whh1T, bih1, bhh1, nullptr, outp);
}
